// MultiheadAttention_16295105921351
// MI455X (gfx1250) — matrix operand order repaired and checked by reading, not compiled, not run on hardware
//
#include <hip/hip_runtime.h>
#include <hip/hip_bf16.h>
#include <math.h>

// Problem constants (match reference)
#define BB 4
#define SS 2048
#define DD 1024
#define HH 16
#define HDD 64        // head dim
#define ROPE_LN_THETA 9.210340371976184f   // ln(10000)

typedef __bf16 bf16_t;
typedef __attribute__((ext_vector_type(16))) __bf16 bf16x16;
typedef __attribute__((ext_vector_type(8)))  __bf16 bf16x8;
typedef __attribute__((ext_vector_type(8)))  float  f32x8;
typedef __attribute__((ext_vector_type(4)))  unsigned int v4u;
typedef __attribute__((ext_vector_type(8)))  int v8i;
typedef __attribute__((ext_vector_type(4)))  int v4i;

union AFragU { bf16x16 v; bf16x8 h[2]; };

// D = A(16x32 bf16) * B(32x16 bf16) + C(16x16 f32)
__device__ __forceinline__ f32x8 wmma_bf(bf16x16 a, bf16x16 b, f32x8 c) {
    return __builtin_amdgcn_wmma_f32_16x16x32_bf16(
        /*neg_a=*/false, a, /*neg_b=*/false, b,
        /*c_mod=*/(short)0, c, /*reuse_a=*/false, /*reuse_b=*/false);
}

// A-fragment (16x32, 16-bit): lane holds row M = lo; K halves split by hi.
// CDNA5 ISA: lanes 0-15 -> K {0..7,16..23}; lanes 16-31 -> K {8..15,24..31}
__device__ __forceinline__ bf16x16 loadA(const bf16_t* base, int row, int ld,
                                         int kbase, int hi) {
    AFragU a;
    const bf16_t* p = base + (size_t)row * ld + kbase + hi * 8;
    a.h[0] = *(const bf16x8*)(p);
    a.h[1] = *(const bf16x8*)(p + 16);
    return a.v;
}

// B-fragment (32x16, 16-bit) from an LDS tile laid out [col][k] with leading
// dim ldk: lane holds column N = lo; K {8*hi..8*hi+7, 16+8*hi..16+8*hi+7}.
__device__ __forceinline__ bf16x16 loadB_lds(const bf16_t* tile, int col,
                                             int ldk, int kbase, int hi) {
    AFragU b;
    const bf16_t* p = tile + col * ldk + kbase + hi * 8;
    b.h[0] = *(const bf16x8*)(p);
    b.h[1] = *(const bf16x8*)(p + 16);
    return b.v;
}

// ---------------------------------------------------------------------------
// Tensor Data Mover: DMA a 2D tile (tile1 rows x tile0 contiguous elements,
// 2-byte elements, row stride stride0 elements) from global into packed LDS.
// D# packing per CDNA5 ISA; groups 2/3 zero (2D tile).
// Tracked on TENSORcnt. EXEC is ignored by tensor ops.
// NOTE: this toolchain exposes the 6-arg builtin
//   (uint32x4 g0, int32x8 g1, int32x4, int32x4, int32x8, i32 cpol).
__device__ __forceinline__ void tdm_load_2d(const bf16_t* gsrc, void* ldst,
                                            unsigned tdim0, unsigned tdim1,
                                            unsigned tile0, unsigned tile1,
                                            unsigned long long stride0) {
    unsigned long long ga = (unsigned long long)(uintptr_t)gsrc;
    unsigned lds_addr = (unsigned)(uintptr_t)ldst;   // low 32 bits = LDS offset
    v4u g0;
    g0[0] = 1u;                                      // count=1, flags=0
    g0[1] = lds_addr;                                // [63:32] lds_addr
    g0[2] = (unsigned)(ga & 0xFFFFFFFFu);            // [95:64] global lo
    g0[3] = (unsigned)((ga >> 32) & 0x01FFFFFFu)     // [120:96] global hi
          | (2u << 30);                              // [127:126] type=2 (image)
    v8i g1;
    g1[0] = (int)(1u << 16);                         // wg_mask=0, data_size=1(2B)
    g1[1] = (int)((tdim0 & 0xFFFFu) << 16);          // abar=0 | tensor_dim0 lo
    g1[2] = (int)(((tdim0 >> 16) & 0xFFFFu) | ((tdim1 & 0xFFFFu) << 16));
    g1[3] = (int)(((tdim1 >> 16) & 0xFFFFu) | ((tile0 & 0xFFFFu) << 16));
    g1[4] = (int)(tile1 & 0xFFFFu);                  // tile_dim1 | tile_dim2=0
    g1[5] = (int)(unsigned)(stride0 & 0xFFFFFFFFu);  // dim0 stride lo
    g1[6] = (int)(unsigned)((stride0 >> 32) & 0xFFFFu); // stride hi | d1str lo=0
    g1[7] = 0;
    v4i z4 = {0, 0, 0, 0};
    v8i z8 = {0, 0, 0, 0, 0, 0, 0, 0};
    __builtin_amdgcn_tensor_load_to_lds(g0, g1, z4, z4, z8, 0);
}

// ---------------------------------------------------------------------------
__global__ void f32_to_bf16_kernel(const float* __restrict__ src,
                                   bf16_t* __restrict__ dst, int n) {
    int i = blockIdx.x * blockDim.x + threadIdx.x;
    if (i < n) dst[i] = (bf16_t)src[i];
}

// ---------------------------------------------------------------------------
// Fused QKV projection + RoPE.  grid = (B*S/64, H, 3), block = 128 (4 waves).
// All 4 waves share one (head, mat) -> the W tile is TDM-staged in LDS once
// per K-step; each wave owns a 16-token M tile (4 C-fragments, 64 outputs).
// Q,K (RoPE'd) -> [B,H,S,HD] bf16 ; V -> transposed [B,H,HD,S] bf16.
__global__ __launch_bounds__(128)
void qkv_rope_kernel(const bf16_t* __restrict__ xbf,
                     const bf16_t* __restrict__ Wqb,
                     const bf16_t* __restrict__ Wkb,
                     const bf16_t* __restrict__ Wvb,
                     const float*  __restrict__ bq,
                     const float*  __restrict__ bk,
                     const float*  __restrict__ bv,
                     bf16_t* __restrict__ Qr,
                     bf16_t* __restrict__ Kr,
                     bf16_t* __restrict__ VT) {
    __shared__ bf16_t Wt[64 * 32];                  // [out-col][k-chunk]
    const int lane = threadIdx.x & 31;
    const int wave = threadIdx.x >> 5;
    const int lo = lane & 15, hi = lane >> 4;
    const int mtile = blockIdx.x * 4 + wave;        // 16-token tile
    const int h     = blockIdx.y;                   // head
    const int mat   = blockIdx.z;                   // 0=Q 1=K 2=V

    const bf16_t* W   = (mat == 0) ? Wqb : (mat == 1) ? Wkb : Wvb;
    const float* bias = (mat == 0) ? bq  : (mat == 1) ? bk  : bv;

    const int arow = mtile * 16 + lo;               // flat token index (B*S)
    f32x8 acc[4] = {};
    for (int kb = 0; kb < DD; kb += 32) {
        __syncthreads();                            // prev-tile LDS reads done
        if (wave == 0) {
            // 64 rows (head outputs) x 32 k, row stride D
            tdm_load_2d(W + (size_t)h * HDD * DD + kb, Wt,
                        DD, DD, 32, 64, DD);
            __builtin_amdgcn_s_wait_tensorcnt(0);
        }
        __syncthreads();
        bf16x16 a = loadA(xbf, arow, DD, kb, hi);
        #pragma unroll
        for (int n = 0; n < 4; ++n)
            acc[n] = wmma_bf(a, loadB_lds(Wt, n * 16 + lo, 32, 0, hi), acc[n]);
    }
    #pragma unroll
    for (int n = 0; n < 4; ++n) {
        float bval = bias[h * HDD + n * 16 + lo];
        #pragma unroll
        for (int r = 0; r < 8; ++r) acc[n][r] += bval;
    }

    if (mat < 2) {
        // RoPE: pair j (frag n) with j+32 (frag n+2); same lane & VGPR slot.
        #pragma unroll
        for (int n = 0; n < 2; ++n) {
            int j = n * 16 + lo;                     // 0..31
            float invf = __expf(-(float)j * (ROPE_LN_THETA / 32.0f));
            #pragma unroll
            for (int r = 0; r < 8; ++r) {
                int tok  = mtile * 16 + r + hi * 8;
                int spos = tok & (SS - 1);
                float ang = (float)spos * invf;
                float c = __cosf(ang), s = __sinf(ang);
                float x1 = acc[n][r], x2 = acc[n + 2][r];
                acc[n][r]     = x1 * c - x2 * s;
                acc[n + 2][r] = x2 * c + x1 * s;
            }
        }
    }

    #pragma unroll
    for (int n = 0; n < 4; ++n) {
        #pragma unroll
        for (int r = 0; r < 8; ++r) {
            int tok  = mtile * 16 + r + hi * 8;
            int b    = tok / SS, spos = tok % SS;
            int j    = n * 16 + lo;
            if (mat == 2) {
                VT[(((size_t)(b * HH + h)) * HDD + j) * SS + spos] =
                    (bf16_t)acc[n][r];
            } else {
                bf16_t* dst = (mat == 0) ? Qr : Kr;
                dst[(((size_t)(b * HH + h)) * SS + spos) * HDD + j] =
                    (bf16_t)acc[n][r];
            }
        }
    }
}

// ---------------------------------------------------------------------------
// Flash-style causal attention. grid = (B*H, S/64), block = 128 (4 waves).
// The block owns a 64-q-row super-tile; all waves share the kb loop so K/V
// 32-wide blocks are TDM-staged into LDS once per block (not once per wave).
__global__ __launch_bounds__(128)
void attn_kernel(const bf16_t* __restrict__ Qr,
                 const bf16_t* __restrict__ Kr,
                 const bf16_t* __restrict__ VT,
                 bf16_t* __restrict__ merged) {
    __shared__ bf16_t Kt[32 * 64];                  // [k-local][d]
    __shared__ bf16_t Vt[64 * 32];                  // [hd][k-local]
    __shared__ bf16_t pT[4][16 * 32];               // per-wave P tile
    const int lane = threadIdx.x & 31;
    const int wave = threadIdx.x >> 5;
    const int lo = lane & 15, hi = lane >> 4;
    const int bh = blockIdx.x;
    const int qt = blockIdx.y * 4 + wave;           // this wave's 16-q tile
    const int b = bh / HH, h = bh % HH;

    const bf16_t* Qb = Qr + (size_t)bh * SS * HDD;
    const bf16_t* Kb = Kr + (size_t)bh * SS * HDD;
    const bf16_t* Vb = VT + (size_t)bh * HDD * SS;

    const int qrow = qt * 16 + lo;
    bf16x16 qa0 = loadA(Qb, qrow, HDD, 0, hi);      // d = 0..31
    bf16x16 qa1 = loadA(Qb, qrow, HDD, 32, hi);     // d = 32..63

    f32x8 o[4] = {};
    float mrun[8], lrun[8];
    #pragma unroll
    for (int r = 0; r < 8; ++r) { mrun[r] = -INFINITY; lrun[r] = 0.0f; }

    const int blockKmax = blockIdx.y * 64 + 63;     // max q row in this block
    for (int kb = 0; kb <= blockKmax; kb += 32) {
        __syncthreads();                            // prev-tile LDS reads done
        if (wave == 0) {
            // K: 32 rows x 64 d (row stride HD); V^T: 64 rows x 32 k (stride S)
            tdm_load_2d(Kb + (size_t)kb * HDD, Kt, HDD, SS, 64, 32, HDD);
            tdm_load_2d(Vb + kb,               Vt, SS, HDD, 32, 64, SS);
            __builtin_amdgcn_s_wait_tensorcnt(0);
        }
        __syncthreads();

        f32x8 sc[2];
        #pragma unroll
        for (int t = 0; t < 2; ++t) {
            const int kpos = kb + t * 16 + lo;      // this lane's k column
            f32x8 s = {};
            s = wmma_bf(qa0, loadB_lds(Kt, t * 16 + lo, 64, 0,  hi), s);
            s = wmma_bf(qa1, loadB_lds(Kt, t * 16 + lo, 64, 32, hi), s);
            #pragma unroll
            for (int r = 0; r < 8; ++r) {
                int q = qt * 16 + r + hi * 8;
                sc[t][r] = (kpos > q) ? -INFINITY : s[r] * 0.125f; // HD^-0.5
            }
        }
        // Online softmax: row stats live across 16-lane halves of each VGPR.
        #pragma unroll
        for (int r = 0; r < 8; ++r) {
            float mx = fmaxf(sc[0][r], sc[1][r]);
            #pragma unroll
            for (int w = 1; w < 16; w <<= 1) mx = fmaxf(mx, __shfl_xor(mx, w, 32));
            float mnew  = fmaxf(mrun[r], mx);
            float alpha = __expf(mrun[r] - mnew);
            float p0 = __expf(sc[0][r] - mnew);
            float p1 = __expf(sc[1][r] - mnew);
            float sum = p0 + p1;
            #pragma unroll
            for (int w = 1; w < 16; w <<= 1) sum += __shfl_xor(sum, w, 32);
            lrun[r] = lrun[r] * alpha + sum;
            mrun[r] = mnew;
            #pragma unroll
            for (int n = 0; n < 4; ++n) o[n][r] *= alpha;
            sc[0][r] = p0; sc[1][r] = p1;
        }
        // C-layout -> A-layout relayout of exp(P) through per-wave LDS tile
        // (DS ops are in-order within a wave; no extra barrier needed).
        #pragma unroll
        for (int t = 0; t < 2; ++t)
            #pragma unroll
            for (int r = 0; r < 8; ++r)
                pT[wave][(r + hi * 8) * 32 + t * 16 + lo] = (bf16_t)sc[t][r];
        AFragU pa;
        pa.h[0] = *(const bf16x8*)&pT[wave][lo * 32 + hi * 8];
        pa.h[1] = *(const bf16x8*)&pT[wave][lo * 32 + 16 + hi * 8];
        // O += P(16x32) @ V(32x64), B-frags from the staged V^T tile.
        #pragma unroll
        for (int n = 0; n < 4; ++n)
            o[n] = wmma_bf(pa.v, loadB_lds(Vt, n * 16 + lo, 32, 0, hi), o[n]);
    }

    #pragma unroll
    for (int r = 0; r < 8; ++r) {
        float inv = 1.0f / lrun[r];
        int spos = qt * 16 + r + hi * 8;
        #pragma unroll
        for (int n = 0; n < 4; ++n) {
            merged[((size_t)(b * SS + spos)) * DD + h * HDD + n * 16 + lo] =
                (bf16_t)(o[n][r] * inv);
        }
    }
}

// ---------------------------------------------------------------------------
// Output projection: out = merged @ Wo^T + bo (fp32 out).
// grid = (B*S/64, D/64), block = 128 (4 waves); Wo tile TDM-staged per K-step.
__global__ __launch_bounds__(128)
void out_proj_kernel(const bf16_t* __restrict__ merged,
                     const bf16_t* __restrict__ Wob,
                     const float*  __restrict__ bo,
                     float* __restrict__ out) {
    __shared__ bf16_t Wt[64 * 32];
    const int lane = threadIdx.x & 31;
    const int wave = threadIdx.x >> 5;
    const int lo = lane & 15, hi = lane >> 4;
    const int mtile = blockIdx.x * 4 + wave;
    const int nt    = blockIdx.y;                   // 64-wide output chunk
    const int arow  = mtile * 16 + lo;

    f32x8 acc[4] = {};
    for (int kb = 0; kb < DD; kb += 32) {
        __syncthreads();
        if (wave == 0) {
            tdm_load_2d(Wob + (size_t)nt * 64 * DD + kb, Wt,
                        DD, DD, 32, 64, DD);
            __builtin_amdgcn_s_wait_tensorcnt(0);
        }
        __syncthreads();
        bf16x16 a = loadA(merged, arow, DD, kb, hi);
        #pragma unroll
        for (int n = 0; n < 4; ++n)
            acc[n] = wmma_bf(a, loadB_lds(Wt, n * 16 + lo, 32, 0, hi), acc[n]);
    }
    #pragma unroll
    for (int n = 0; n < 4; ++n) {
        int col = nt * 64 + n * 16 + lo;
        float bval = bo[col];
        #pragma unroll
        for (int r = 0; r < 8; ++r) {
            int tok = mtile * 16 + r + hi * 8;
            out[(size_t)tok * DD + col] = acc[n][r] + bval;
        }
    }
}

// ---------------------------------------------------------------------------
extern "C" void kernel_launch(void* const* d_in, const int* in_sizes, int n_in,
                              void* d_out, int out_size, void* d_ws, size_t ws_size,
                              hipStream_t stream) {
    const float* x  = (const float*)d_in[0];
    const float* Wq = (const float*)d_in[1];
    const float* bq = (const float*)d_in[2];
    const float* Wk = (const float*)d_in[3];
    const float* bk = (const float*)d_in[4];
    const float* Wv = (const float*)d_in[5];
    const float* bv = (const float*)d_in[6];
    const float* Wo = (const float*)d_in[7];
    const float* bo = (const float*)d_in[8];
    float* out = (float*)d_out;

    char* ws = (char*)d_ws;
    size_t off = 0;
    const size_t nTok = (size_t)BB * SS;            // 8192 tokens
    bf16_t* xbf    = (bf16_t*)(ws + off); off += nTok * DD * sizeof(bf16_t);
    bf16_t* wqb    = (bf16_t*)(ws + off); off += (size_t)DD * DD * sizeof(bf16_t);
    bf16_t* wkb    = (bf16_t*)(ws + off); off += (size_t)DD * DD * sizeof(bf16_t);
    bf16_t* wvb    = (bf16_t*)(ws + off); off += (size_t)DD * DD * sizeof(bf16_t);
    bf16_t* wob    = (bf16_t*)(ws + off); off += (size_t)DD * DD * sizeof(bf16_t);
    bf16_t* Qr     = (bf16_t*)(ws + off); off += nTok * DD * sizeof(bf16_t);
    bf16_t* Kr     = (bf16_t*)(ws + off); off += nTok * DD * sizeof(bf16_t);
    bf16_t* VT     = (bf16_t*)(ws + off); off += nTok * DD * sizeof(bf16_t);
    bf16_t* merged = (bf16_t*)(ws + off); off += nTok * DD * sizeof(bf16_t);

    const int nx = BB * SS * DD;
    const int nw = DD * DD;
    f32_to_bf16_kernel<<<(nx + 255) / 256, 256, 0, stream>>>(x,  xbf, nx);
    f32_to_bf16_kernel<<<(nw + 255) / 256, 256, 0, stream>>>(Wq, wqb, nw);
    f32_to_bf16_kernel<<<(nw + 255) / 256, 256, 0, stream>>>(Wk, wkb, nw);
    f32_to_bf16_kernel<<<(nw + 255) / 256, 256, 0, stream>>>(Wv, wvb, nw);
    f32_to_bf16_kernel<<<(nw + 255) / 256, 256, 0, stream>>>(Wo, wob, nw);

    dim3 g1(BB * SS / 64, HH, 3);
    qkv_rope_kernel<<<g1, 128, 0, stream>>>(xbf, wqb, wkb, wvb, bq, bk, bv,
                                            Qr, Kr, VT);
    dim3 g2(BB * HH, SS / 64);
    attn_kernel<<<g2, 128, 0, stream>>>(Qr, Kr, VT, merged);

    dim3 g3(BB * SS / 64, DD / 64);
    out_proj_kernel<<<g3, 128, 0, stream>>>(merged, wob, bo, out);
}
